// GNN_37409165149000
// MI455X (gfx1250) — matrix operand order repaired and checked by reading, not compiled, not run on hardware
//
#include <hip/hip_runtime.h>
#include <hip/hip_bf16.h>

typedef __attribute__((ext_vector_type(16))) __bf16 v16bf;
typedef __attribute__((ext_vector_type(8)))  __bf16 v8bf;
typedef __attribute__((ext_vector_type(8)))  float  v8f;

#define N_NODES   100000
#define N_EDGES   625000
#define D         128
#define D_OUT     10
#define NUM_GRAPHS 256
#define ROW_TILES (N_NODES / 16)   /* 6250, exact */

// ---------------------------------------------------------------------------
// f32 -> bf16 conversion (features and weights)
// ---------------------------------------------------------------------------
__global__ void cvt_bf16(const float* __restrict__ in, __bf16* __restrict__ out, int n) {
    int i = blockIdx.x * blockDim.x + threadIdx.x;
    if (i < n) out[i] = (__bf16)in[i];
}

__global__ void zero_f32(float* __restrict__ p, int n) {
    int i = blockIdx.x * blockDim.x + threadIdx.x;
    if (i < n) p[i] = 0.0f;
}

// ---------------------------------------------------------------------------
// Fused dual NT GEMM: Out1 = X*W1^T, Out2 = X*W2^T  (all [*,128], K contig).
// Both 128x128 bf16 weight matrices staged in LDS once per block (64 KB of
// the 320 KB WGP LDS); B fragments then come from ds_load_b128 instead of
// per-WMMA global loads. A fragments of X are loaded once and reused for
// both outputs. bf16 in / f32 accumulate via v_wmma_f32_16x16x32_bf16.
// One wave per 16-row strip; 2 x 8 N-tiles x 4 K-steps = 64 WMMAs per wave.
// ---------------------------------------------------------------------------
__global__ void __launch_bounds__(256)
gemm_dual_128(const __bf16* __restrict__ X,
              const __bf16* __restrict__ W1, const __bf16* __restrict__ W2,
              float* __restrict__ Out1, float* __restrict__ Out2) {
    __shared__ __bf16 ldsW[2][D * D];          // 2 x 32 KB

    // --- stage both weight matrices into LDS (all 256 threads) ---
    {
        const uint4* s1 = (const uint4*)W1;
        const uint4* s2 = (const uint4*)W2;
        uint4* d1 = (uint4*)ldsW[0];
        uint4* d2 = (uint4*)ldsW[1];
        const int nvec = D * D * 2 / 16;       // 2048 16B vectors per matrix
        for (int i = threadIdx.x; i < nvec; i += 256) {
            d1[i] = s1[i];
            d2[i] = s2[i];
        }
    }
    __syncthreads();

    const int lane = threadIdx.x & 31;
    const int wave = threadIdx.x >> 5;
    const int rowTile = blockIdx.x * 8 + wave;
    if (rowTile >= ROW_TILES) return;          // wave-uniform: EXEC stays all-1s

    const int m0   = rowTile * 16;
    const int lo16 = lane & 15;
    const int hiA  = (lane >= 16) ? 8  : 0;    // unused; A also loads at k0 + hiB
    const int hiB  = (lane >= 16) ? 16 : 0;    // lanes 16-31 take k0+16..31 (contiguous k, A and B)

    v8f acc1[8] = {};
    v8f acc2[8] = {};

    const __bf16* arow = X + (size_t)(m0 + lo16) * D;
#pragma unroll
    for (int k0 = 0; k0 < D; k0 += 32) {
        // A fragment: row m0+lo16, 16 contig halves {k0+hiB..+15}
        v16bf a = *(const v16bf*)(arow + k0 + hiB);

#pragma unroll
        for (int nb = 0; nb < 8; nb++) {
            const int woff = (nb * 16 + lo16) * D + k0 + hiB;  // 16 contig halves
            v16bf b1 = *(const v16bf*)(ldsW[0] + woff);
            v16bf b2 = *(const v16bf*)(ldsW[1] + woff);
            acc1[nb] = __builtin_amdgcn_wmma_f32_16x16x32_bf16(
                false, a, false, b1, (short)0, acc1[nb], false, false);
            acc2[nb] = __builtin_amdgcn_wmma_f32_16x16x32_bf16(
                false, a, false, b2, (short)0, acc2[nb], false, false);
        }
    }

    // C/D layout: VGPR r -> row m0 + r + (lane>=16 ? 8 : 0), col = nb*16 + lo16
    const int rbase = m0 + ((lane >= 16) ? 8 : 0);
#pragma unroll
    for (int nb = 0; nb < 8; nb++) {
#pragma unroll
        for (int r = 0; r < 8; r++) {
            const size_t o = (size_t)(rbase + r) * D + nb * 16 + lo16;
            Out1[o] = acc1[nb][r];
            Out2[o] = acc2[nb][r];
        }
    }
}

// ---------------------------------------------------------------------------
// Edge scatter-add: agg[row[e]] += nbr[col[e]].  One wave per edge, one
// float4 per lane (32*16B = full 512B row, perfectly coalesced gather).
// ---------------------------------------------------------------------------
__global__ void __launch_bounds__(256)
edge_scatter(const float* __restrict__ nbr, const int* __restrict__ row,
             const int* __restrict__ col, float* __restrict__ agg) {
    int t = blockIdx.x * blockDim.x + threadIdx.x;
    int e = t >> 5;
    if (e >= N_EDGES) return;
    int lane = t & 31;
    int r = row[e], c = col[e];
    const float4 v = ((const float4*)(nbr + (size_t)c * D))[lane];
    float* dst = agg + (size_t)r * D + lane * 4;
    atomicAdd(dst + 0, v.x);
    atomicAdd(dst + 1, v.y);
    atomicAdd(dst + 2, v.z);
    atomicAdd(dst + 3, v.w);
}

// ---------------------------------------------------------------------------
// Fused ReLU (in place, f32 h) + bf16 conversion for next layer's WMMA input
// ---------------------------------------------------------------------------
__global__ void relu_cvt(float* __restrict__ h, __bf16* __restrict__ xb, int n) {
    int i = blockIdx.x * blockDim.x + threadIdx.x;
    if (i < n) {
        float v = h[i];
        v = v > 0.0f ? v : 0.0f;
        h[i] = v;
        xb[i] = (__bf16)v;
    }
}

// ---------------------------------------------------------------------------
// Per-graph sum pool (+counts). One wave per node, float4 per lane.
// ---------------------------------------------------------------------------
__global__ void __launch_bounds__(256)
pool_sum(const float* __restrict__ h, const int* __restrict__ batch,
         float* __restrict__ pooled, float* __restrict__ counts) {
    int t = blockIdx.x * blockDim.x + threadIdx.x;
    int node = t >> 5;
    if (node >= N_NODES) return;
    int lane = t & 31;
    int g = batch[node];
    const float4 v = ((const float4*)(h + (size_t)node * D))[lane];
    float* dst = pooled + (size_t)g * D + lane * 4;
    atomicAdd(dst + 0, v.x);
    atomicAdd(dst + 1, v.y);
    atomicAdd(dst + 2, v.z);
    atomicAdd(dst + 3, v.w);
    if (lane == 0) atomicAdd(counts + g, 1.0f);
}

// ---------------------------------------------------------------------------
// Classifier: out[g,o] = (pooled[g]·Wc[o]) / max(cnt,1) + bc[o].  2560 dots
// of length 128 — scalar f32 is fine (negligible work).
// ---------------------------------------------------------------------------
__global__ void classify(const float* __restrict__ pooled, const float* __restrict__ counts,
                         const float* __restrict__ Wc, const float* __restrict__ bc,
                         float* __restrict__ out) {
    int t = blockIdx.x * blockDim.x + threadIdx.x;
    if (t >= NUM_GRAPHS * D_OUT) return;
    int g = t / D_OUT, o = t % D_OUT;
    float inv = 1.0f / fmaxf(counts[g], 1.0f);
    const float* pr = pooled + (size_t)g * D;
    const float* wr = Wc + (size_t)o * D;
    float s = 0.0f;
#pragma unroll 8
    for (int k = 0; k < D; k++) s += pr[k] * wr[k];
    out[t] = s * inv + bc[o];
}

// ---------------------------------------------------------------------------
extern "C" void kernel_launch(void* const* d_in, const int* in_sizes, int n_in,
                              void* d_out, int out_size, void* d_ws, size_t ws_size,
                              hipStream_t stream) {
    const float* x     = (const float*)d_in[0];
    const int*   eidx  = (const int*)d_in[1];
    const int*   batch = (const int*)d_in[2];
    const float* Wf[6] = { (const float*)d_in[3], (const float*)d_in[4],
                           (const float*)d_in[5], (const float*)d_in[6],
                           (const float*)d_in[7], (const float*)d_in[8] };
    const float* Wc = (const float*)d_in[9];
    const float* bc = (const float*)d_in[10];
    float* out = (float*)d_out;

    // Workspace carve-out (256B aligned slices), ~128.5 MB total
    char* ws = (char*)d_ws;
    size_t off = 0;
    auto carve = [&](size_t bytes) -> char* {
        char* p = ws + off;
        off = (off + bytes + 255) & ~(size_t)255;
        return p;
    };
    __bf16* Xb     = (__bf16*)carve((size_t)N_NODES * D * 2);      // bf16 features
    float*  SelfB  = (float*) carve((size_t)N_NODES * D * 4);      // self term / agg / h
    float*  NbrB   = (float*) carve((size_t)N_NODES * D * 4);      // x @ W2^T
    __bf16* Wb     = (__bf16*)carve((size_t)6 * D * D * 2);        // bf16 weights
    float*  pooled = (float*) carve((size_t)NUM_GRAPHS * D * 4);
    float*  counts = (float*) carve((size_t)NUM_GRAPHS * 4);

    const int* row = eidx;            // edge_index[0, :]
    const int* col = eidx + N_EDGES;  // edge_index[1, :]

    const int nfeat = N_NODES * D;
    cvt_bf16<<<(nfeat + 255) / 256, 256, 0, stream>>>(x, Xb, nfeat);
    for (int i = 0; i < 6; i++)
        cvt_bf16<<<(D * D + 255) / 256, 256, 0, stream>>>(Wf[i], Wb + (size_t)i * D * D, D * D);

    const int gemmBlocks = (ROW_TILES + 7) / 8;     // 8 waves (16-row strips) per block
    for (int l = 0; l < 3; l++) {
        gemm_dual_128<<<gemmBlocks, 256, 0, stream>>>(
            Xb, Wb + (size_t)(2 * l) * D * D, Wb + (size_t)(2 * l + 1) * D * D,
            SelfB, NbrB);
        edge_scatter<<<(N_EDGES * 32 + 255) / 256, 256, 0, stream>>>(NbrB, row, col, SelfB);
        relu_cvt<<<(nfeat + 255) / 256, 256, 0, stream>>>(SelfB, Xb, nfeat);
    }

    zero_f32<<<(NUM_GRAPHS * D + 255) / 256, 256, 0, stream>>>(pooled, NUM_GRAPHS * D);
    zero_f32<<<1, 256, 0, stream>>>(counts, NUM_GRAPHS);
    pool_sum<<<(N_NODES * 32 + 255) / 256, 256, 0, stream>>>(SelfB, batch, pooled, counts);
    classify<<<(NUM_GRAPHS * D_OUT + 255) / 256, 256, 0, stream>>>(pooled, counts, Wc, bc, out);
}
